// Lightning_66915590471926
// MI455X (gfx1250) — compile-verified, matrix operand order checked by reading, not run on hardware
//
#include <hip/hip_runtime.h>
#include <math.h>

// ---------------------------------------------------------------------------
// CDNA5 (gfx1250) implementation, round 3.
//  * One-time f32->f16 conversion of x and all big weights; GEMMs read/write
//    f16 (halves HBM traffic, feeds v_wmma_f32_16x16x32_f16 directly).
//  * Double-buffered LDS staging via GLOBAL_LOAD_ASYNC_TO_LDS_B128 +
//    s_wait_asynccnt (ASYNCcnt pipeline), sync fallback otherwise.
//  * Attention dead-code-eliminated to the 512 q-rows / V[:,511] that feed
//    the single surviving output row.
// ---------------------------------------------------------------------------

typedef _Float16 v16h __attribute__((ext_vector_type(16)));
typedef _Float16 h8   __attribute__((ext_vector_type(8)));
typedef float    v8f  __attribute__((ext_vector_type(8)));

#define BM 64
#define BN 128
#define BK 32
#define LDT 40   // padded LDS row stride in halves (80B, 16B-aligned rows)

#if defined(__HIP_DEVICE_COMPILE__) && \
    __has_builtin(__builtin_amdgcn_global_load_async_to_lds_b128) && \
    __has_builtin(__builtin_amdgcn_s_wait_asynccnt)
#define USE_ASYNC 1
#else
#define USE_ASYNC 0
#endif

// builtin takes pointers to GCC-style int vectors (vector_size), AS1/AS3
typedef int b128i __attribute__((vector_size(16)));
typedef __attribute__((address_space(1))) b128i gb128;
typedef __attribute__((address_space(3))) b128i lb128;

// copy 16 bytes (8 halves) global -> LDS
__device__ __forceinline__ void copy16(const _Float16* g, _Float16* l)
{
#if USE_ASYNC
    __builtin_amdgcn_global_load_async_to_lds_b128((gb128*)g, (lb128*)l, 0, 0);
#else
    *(h8*)l = *(const h8*)g;
#endif
}

template <int N> __device__ __forceinline__ void async_wait()
{
#if USE_ASYNC
    __builtin_amdgcn_s_wait_asynccnt(N);
#endif
}

// C = act(scale * A(MxK) @ B(NxK)^T + bias); A,B f16; C f16 or f32.
__global__ __launch_bounds__(256) void gemm_wmma_f16(
    const _Float16* __restrict__ A, int lda,
    const _Float16* __restrict__ B, int ldb,
    const float* __restrict__ bias,
    void* __restrict__ C, int ldc,
    int K, int relu, int out_f32, float scale)
{
    __shared__ _Float16 As[2][BM * LDT];
    __shared__ _Float16 Bs[2][BN * LDT];

    const int tid  = threadIdx.x;
    const int lane = tid & 31;
    const int wave = tid >> 5;
    const int wm   = wave >> 2;   // 0..1
    const int wn   = wave & 3;    // 0..3
    const int bm   = blockIdx.y * BM;
    const int bn   = blockIdx.x * BN;

    // staging coords: A tile 64x32 halves = 1 b128/thread; B 128x32 = 2.
    const int ar = tid >> 2;            // 0..63
    const int ac = (tid & 3) * 8;       // 0,8,16,24 halves
    const int br = tid >> 1;            // 0..127
    const int bc = (tid & 1) * 16;      // 0,16 halves

    const _Float16* Ag = A + (size_t)(bm + ar) * lda + ac;
    const _Float16* Bg = B + (size_t)(bn + br) * ldb + bc;

    auto stage = [&](int buf, int k0) {
        copy16(Ag + k0, &As[buf][ar * LDT + ac]);
        copy16(Bg + k0,     &Bs[buf][br * LDT + bc]);
        copy16(Bg + k0 + 8, &Bs[buf][br * LDT + bc + 8]);
    };

    v8f acc[2][2] = {};
    const int half = lane >> 4;
    const int mr   = lane & 15;
    const int nk   = K / BK;

    stage(0, 0);

    for (int kk = 0; kk < nk; ++kk) {
        const int cur = kk & 1;
        if (kk + 1 < nk) { stage(cur ^ 1, (kk + 1) * BK); async_wait<3>(); }
        else             { async_wait<0>(); }
        __syncthreads();

        // 16-bit A/B fragment layout: half=0 -> K [0,8)+[16,24); half=1 -> [8,16)+[24,32)
        v16h afrag[2], bfrag[2];
        #pragma unroll
        for (int t = 0; t < 2; ++t) {
            const _Float16* ap = &As[cur][(wm * 32 + t * 16 + mr) * LDT];
            h8 lo = *(const h8*)(ap + half * 8);
            h8 hi = *(const h8*)(ap + 16 + half * 8);
            #pragma unroll
            for (int i = 0; i < 8; ++i) { afrag[t][i] = lo[i]; afrag[t][8 + i] = hi[i]; }
        }
        #pragma unroll
        for (int u = 0; u < 2; ++u) {
            const _Float16* bp = &Bs[cur][(wn * 32 + u * 16 + mr) * LDT];
            h8 lo = *(const h8*)(bp + half * 8);
            h8 hi = *(const h8*)(bp + 16 + half * 8);
            #pragma unroll
            for (int i = 0; i < 8; ++i) { bfrag[u][i] = lo[i]; bfrag[u][8 + i] = hi[i]; }
        }

        #pragma unroll
        for (int t = 0; t < 2; ++t)
            #pragma unroll
            for (int u = 0; u < 2; ++u)
                acc[t][u] = __builtin_amdgcn_wmma_f32_16x16x32_f16(
                    false, afrag[t], false, bfrag[u],
                    (short)0, acc[t][u], false, false);
        __syncthreads();
    }

    // epilogue: C/D layout row = r + 8*half, col = lane&15
    #pragma unroll
    for (int t = 0; t < 2; ++t) {
        #pragma unroll
        for (int u = 0; u < 2; ++u) {
            const int col = bn + wn * 32 + u * 16 + mr;
            const float bv = bias ? bias[col] : 0.0f;
            #pragma unroll
            for (int r = 0; r < 8; ++r) {
                const int row = bm + wm * 32 + t * 16 + half * 8 + r;
                float v = acc[t][u][r] * scale + bv;
                if (relu) v = fmaxf(v, 0.0f);
                if (out_f32) ((float*)C)[(size_t)row * ldc + col] = v;
                else         ((_Float16*)C)[(size_t)row * ldc + col] = (_Float16)v;
            }
        }
    }
}

// elementwise f32 -> f16 (n multiple of 4)
__global__ __launch_bounds__(256) void cvt_f16_kernel(const float* __restrict__ in,
                                                      _Float16* __restrict__ out, int n)
{
    const int i = (blockIdx.x * 256 + threadIdx.x) * 4;
    if (i >= n) return;
    float4 f = *(const float4*)(in + i);
    out[i + 0] = (_Float16)f.x; out[i + 1] = (_Float16)f.y;
    out[i + 2] = (_Float16)f.z; out[i + 3] = (_Float16)f.w;
}

// h[s][e] += positional encoding (f16 storage), S=4096, E=512
__global__ __launch_bounds__(256) void posadd_kernel(_Float16* __restrict__ h)
{
    const int idx = blockIdx.x * 256 + threadIdx.x;
    const int s = idx >> 9;
    const int e = idx & 511;
    const int i = e >> 1;
    const float div = __expf((float)(2 * i) * (-9.210340371976184f / 512.0f));
    const float ang = (float)s * div;
    const float pe = (e & 1) ? __cosf(ang) : __sinf(ang);
    h[idx] = (_Float16)((float)h[idx] + pe);
}

// in-place f32 row softmax, one block per row
__global__ __launch_bounds__(256) void softmax_rows(float* __restrict__ S, int cols)
{
    __shared__ float red[256];
    float* row = S + (size_t)blockIdx.x * cols;
    const int t = threadIdx.x;
    float m = -1e30f;
    for (int j = t; j < cols; j += 256) m = fmaxf(m, row[j]);
    red[t] = m; __syncthreads();
    for (int s = 128; s > 0; s >>= 1) { if (t < s) red[t] = fmaxf(red[t], red[t + s]); __syncthreads(); }
    m = red[0]; __syncthreads();
    float sum = 0.0f;
    for (int j = t; j < cols; j += 256) { float e = __expf(row[j] - m); row[j] = e; sum += e; }
    red[t] = sum; __syncthreads();
    for (int s = 128; s > 0; s >>= 1) { if (t < s) red[t] += red[t + s]; __syncthreads(); }
    const float inv = 1.0f / red[0];
    for (int j = t; j < cols; j += 256) row[j] *= inv;
}

// out[r] = sum_j (A1[r][j] - lam*A2[r][j] + (j > 3584+r)) * V[j][511]
__global__ __launch_bounds__(256) void attn_col_kernel(
    const float* __restrict__ A1, const float* __restrict__ A2,
    const _Float16* __restrict__ qkvh, const float* __restrict__ lam_p,
    float* __restrict__ out)
{
    __shared__ float red[256];
    const int r = blockIdx.x;            // 0..511
    const int grow = 3584 + r;
    const float lam = *lam_p;
    const int t = threadIdx.x;
    const float* a1 = A1 + (size_t)r * 4096;
    const float* a2 = A2 + (size_t)r * 4096;
    float acc = 0.0f;
    for (int j = t; j < 4096; j += 256) {
        const float v = (float)qkvh[(size_t)j * 1536 + 1535];   // V column 511
        const float w = a1[j] - lam * a2[j] + ((j > grow) ? 1.0f : 0.0f);
        acc += w * v;
    }
    red[t] = acc; __syncthreads();
    for (int s = 128; s > 0; s >>= 1) { if (t < s) red[t] += red[t + s]; __syncthreads(); }
    if (t == 0) out[r] = red[0];
}

// y[o] = act(x . W[o][:] + b[o]); W f32 NxK row-major (tiny tail)
__global__ void gemv_kernel(const float* __restrict__ x, const float* __restrict__ W,
                            const float* __restrict__ b, float* __restrict__ y,
                            int N, int K, int relu)
{
    const int o = blockIdx.x * blockDim.x + threadIdx.x;
    if (o >= N) return;
    const float* w = W + (size_t)o * K;
    float acc = b ? b[o] : 0.0f;
    for (int k = 0; k < K; ++k) acc += x[k] * w[k];
    if (relu) acc = fmaxf(acc, 0.0f);
    y[o] = acc;
}

__global__ __launch_bounds__(256) void softmax_out_kernel(const float* __restrict__ z,
                                                          float* __restrict__ out)
{
    __shared__ float red[256];
    const int t = threadIdx.x;
    const float v = z[t];
    red[t] = v; __syncthreads();
    for (int s = 128; s > 0; s >>= 1) { if (t < s) red[t] = fmaxf(red[t], red[t + s]); __syncthreads(); }
    const float m = red[0]; __syncthreads();
    const float e = __expf(v - m);
    red[t] = e; __syncthreads();
    for (int s = 128; s > 0; s >>= 1) { if (t < s) red[t] += red[t + s]; __syncthreads(); }
    out[t] = e / red[0];
}

extern "C" void kernel_launch(void* const* d_in, const int* in_sizes, int n_in,
                              void* d_out, int out_size, void* d_ws, size_t ws_size,
                              hipStream_t stream)
{
    const float* x   = (const float*)d_in[0];
    const float* ew1 = (const float*)d_in[1];
    const float* eb1 = (const float*)d_in[2];
    const float* ew2 = (const float*)d_in[3];
    const float* eb2 = (const float*)d_in[4];
    const float* ew3 = (const float*)d_in[5];
    const float* eb3 = (const float*)d_in[6];
    const float* ew4 = (const float*)d_in[7];
    const float* eb4 = (const float*)d_in[8];
    const float* ipw = (const float*)d_in[9];
    const float* ipb = (const float*)d_in[10];
    const float* opw = (const float*)d_in[11];
    const float* opb = (const float*)d_in[12];
    const float* lam = (const float*)d_in[13];
    const float* aw1 = (const float*)d_in[14];
    const float* ab1 = (const float*)d_in[15];
    const float* aw2 = (const float*)d_in[16];
    const float* ab2 = (const float*)d_in[17];
    const float* aw3 = (const float*)d_in[18];
    const float* ab3 = (const float*)d_in[19];

    // ---- workspace layout (halves) ----
    _Float16* ws16 = (_Float16*)d_ws;
    _Float16* xh  = ws16;                    // 4096*1024
    _Float16* wh1 = xh  + (size_t)4194304;   // 2048*1024
    _Float16* wh2 = wh1 + (size_t)2097152;   // 4096*2048
    _Float16* wh3 = wh2 + (size_t)8388608;   // 2048*4096
    _Float16* wh4 = wh3 + (size_t)8388608;   // 512*2048
    _Float16* wih = wh4 + (size_t)1048576;   // 1536*512
    _Float16* h1h = wih + (size_t)786432;    // 4096*2048 (reused: h3h, then s1/s2 f32)
    _Float16* h2h = h1h + (size_t)8388608;   // 4096*4096 (reused: qkvh)
    _Float16* hbh = h2h + (size_t)16777216;  // 4096*512
    float* tail   = (float*)(hbh + (size_t)2097152);
    float* acolv = tail;          // 512
    float* yv    = acolv + 512;   // 512
    float* z1    = yv + 512;      // 1024
    float* z2    = z1 + 1024;     // 2048
    float* z3    = z2 + 2048;     // 256

    _Float16* h3h  = h1h;                 // reuse
    _Float16* qkvh = h2h;                 // reuse
    float* s1 = (float*)h1h;              // reuse (512x4096 f32)
    float* s2 = s1 + (size_t)512 * 4096;  // fits exactly in the h1h region

    const dim3 blk(256);
    auto grid = [](int M, int N) { return dim3(N / BN, M / BM); };

    // one-time f32 -> f16 conversion of x and the big weights
    cvt_f16_kernel<<<4194304 / 1024, blk, 0, stream>>>(x,   xh,  4194304);
    cvt_f16_kernel<<<2097152 / 1024, blk, 0, stream>>>(ew1, wh1, 2097152);
    cvt_f16_kernel<<<8388608 / 1024, blk, 0, stream>>>(ew2, wh2, 8388608);
    cvt_f16_kernel<<<8388608 / 1024, blk, 0, stream>>>(ew3, wh3, 8388608);
    cvt_f16_kernel<<<1048576 / 1024, blk, 0, stream>>>(ew4, wh4, 1048576);
    cvt_f16_kernel<<< 786432 / 1024, blk, 0, stream>>>(ipw, wih, 786432);

    // Encoder MLP (dominant ~163 GFLOP), all-f16 storage, f32 accum
    gemm_wmma_f16<<<grid(4096, 2048), blk, 0, stream>>>(xh, 1024,  wh1, 1024, eb1, h1h, 2048, 1024, 1, 0, 1.0f);
    gemm_wmma_f16<<<grid(4096, 4096), blk, 0, stream>>>(h1h, 2048, wh2, 2048, eb2, h2h, 4096, 2048, 1, 0, 1.0f);
    gemm_wmma_f16<<<grid(4096, 2048), blk, 0, stream>>>(h2h, 4096, wh3, 4096, eb3, h3h, 2048, 4096, 1, 0, 1.0f);
    gemm_wmma_f16<<<grid(4096,  512), blk, 0, stream>>>(h3h, 2048, wh4, 2048, eb4, hbh,  512, 2048, 0, 0, 1.0f);

    posadd_kernel<<<(4096 * 512) / 256, blk, 0, stream>>>(hbh);

    // QKV projection (packed 4096 x 1536, f16)
    gemm_wmma_f16<<<grid(4096, 1536), blk, 0, stream>>>(hbh, 512, wih, 512, ipb, qkvh, 1536, 512, 0, 0, 1.0f);

    // Scores only for q-rows 3584..4095 (the rows feeding the output), f32 out
    const float scale = 0.70710678118654752f;   // 1/sqrt(HEAD_DIM=2)
    const _Float16* q1 = qkvh + (size_t)3584 * 1536;
    const _Float16* q2 = q1 + 256;
    const _Float16* k1 = qkvh + 512;
    const _Float16* k2 = qkvh + 768;
    gemm_wmma_f16<<<grid(512, 4096), blk, 0, stream>>>(q1, 1536, k1, 1536, nullptr, s1, 4096, 256, 0, 1, scale);
    gemm_wmma_f16<<<grid(512, 4096), blk, 0, stream>>>(q2, 1536, k2, 1536, nullptr, s2, 4096, 256, 0, 1, scale);

    // Row softmax over both heads (s2 contiguous after s1 -> 1024 rows)
    softmax_rows<<<1024, blk, 0, stream>>>(s1, 4096);

    // (a1 - lam*a2 + suffix-mask) . V[:,511]  -> scrambled last row (512)
    attn_col_kernel<<<512, blk, 0, stream>>>(s1, s2, qkvh, lam, acolv);

    // Tail on the single surviving row
    gemv_kernel<<<2, blk, 0, stream>>>(acolv, opw, opb, yv, 512, 512, 0);
    gemv_kernel<<<4, blk, 0, stream>>>(yv, aw1, ab1, z1, 1024, 512, 1);
    gemv_kernel<<<8, blk, 0, stream>>>(z1, aw2, ab2, z2, 2048, 1024, 1);
    gemv_kernel<<<1, blk, 0, stream>>>(z2, aw3, ab3, z3, 256, 2048, 0);
    softmax_out_kernel<<<1, blk, 0, stream>>>(z3, (float*)d_out);
}
